// Decoder_77051713290327
// MI455X (gfx1250) — matrix operand order repaired and checked by reading, not compiled, not run on hardware
//
#include <hip/hip_runtime.h>
#include <hip/hip_bf16.h>
#include <stdint.h>

// ---------------- problem constants ----------------
#define UNITS 128
#define BATCH 256
#define TSTEPS 1024
#define FEAT 128
#define NGATE 512          // 4*UNITS
#define KCAT 256           // FEAT + UNITS (concat [x|h])
#define ROWS 16            // batch rows per workgroup (== WMMA M)
#define NWG (BATCH / ROWS) // 16 workgroups, zero inter-WG sync
#define KP 264             // padded k-pitch (bf16 elems) of transposed weights
#define ZP 520             // padded row pitch (floats) of z scratch

typedef __attribute__((ext_vector_type(16))) __bf16 v16bf;
typedef __attribute__((ext_vector_type(8)))  float  v8f;

union FragU { uint4 q[2]; v16bf v; };
union Pack8 { __bf16 h[8]; uint4 q; };

__device__ __forceinline__ float sigf(float x) {
    return 1.0f / (1.0f + __expf(-x));
}
__device__ __forceinline__ float tanh_fast(float x) {
    // tanh(x) = 2*sigmoid(2x) - 1
    return 2.0f * sigf(2.0f * x) - 1.0f;
}

// A-matrix (16x32 bf16) fragment layout as used here (contiguous k = 16*hi + s, same order as B):
//   lane = m + 16*hi, VGPR j holds K pair; an 8-aligned K-chunk of one row
//   maps to one contiguous 16B chunk of one lane's 32B fragment slice.
// A staging buffer layout: [kk(8)][lane(32)][2 x uint4].
__device__ __forceinline__ int a_chunk_index(int m, int k0) {
    int kk   = k0 >> 5;
    int ko   = k0 & 31;
    int hi   = (ko >> 4) & 1;
    int lane = m + (hi << 4);
    int sel  = (ko >> 3) & 1;
    return (kk * 32 + lane) * 2 + sel;
}

// One [16 x 512] = [16 x KCAT] @ [KCAT x 512] GEMM, 8 waves x 4 N-tiles.
__device__ __forceinline__ void do_gemm(const uint4* __restrict__ A,
                                        const __bf16* __restrict__ Wlds,
                                        float* __restrict__ zlds,
                                        int wave, int lane)
{
    const int nl = lane & 15;
    const int hi = lane >> 4;

    v8f acc[4];
#pragma unroll
    for (int tt = 0; tt < 4; ++tt)
#pragma unroll
        for (int r = 0; r < 8; ++r) acc[tt][r] = 0.0f;

#pragma unroll
    for (int kk = 0; kk < 8; ++kk) {
        FragU fa;
        fa.q[0] = A[(kk * 32 + lane) * 2 + 0];
        fa.q[1] = A[(kk * 32 + lane) * 2 + 1];
#pragma unroll
        for (int tt = 0; tt < 4; ++tt) {
            const int n = wave * 64 + tt * 16 + nl;
            // B fragment: column n, K = kk*32 + hi*16 .. +15  (32B contiguous)
            const uint4* bp =
                (const uint4*)(Wlds + n * KP + kk * 32 + hi * 16);
            FragU fb;
            fb.q[0] = bp[0];
            fb.q[1] = bp[1];
            acc[tt] = __builtin_amdgcn_wmma_f32_16x16x32_bf16(
                false, fa.v, false, fb.v, (short)0, acc[tt], false, false);
        }
    }

    // D layout: VGPR r -> M = r + 8*hi, N = tile_n + nl
#pragma unroll
    for (int tt = 0; tt < 4; ++tt) {
        const int n = wave * 64 + tt * 16 + nl;
#pragma unroll
        for (int r = 0; r < 8; ++r) {
            zlds[(r + hi * 8) * ZP + n] = acc[tt][r];
        }
    }
}

__global__ __launch_bounds__(256, 1)
void lstm_stack2_kernel(const float* __restrict__ x,
                        const float* __restrict__ W,
                        const float* __restrict__ U,
                        const float* __restrict__ b,
                        float* __restrict__ out)
{
    __shared__ __align__(16) __bf16 Wlds[NGATE * KP]; // transposed [n][k], k=0..127:W, 128..255:U
    __shared__ uint4  A1[8 * 32 * 2];                 // cell1 A fragments [x | h1]
    __shared__ uint4  A2[8 * 32 * 2];                 // cell2 A fragments [h1 | h2]
    __shared__ float  zlds[ROWS * ZP];                // z scratch [16 x 512] padded
    __shared__ float  blds[NGATE];

    const int tid  = threadIdx.x;
    const int wg   = blockIdx.x;
    const int lane = tid & 31;
    const int wave = tid >> 5;

    // ---- preload weights transposed to bf16 (one-time, L2-served) ----
    for (int idx = tid; idx < 2 * FEAT * NGATE; idx += 256) {
        const int kAll = idx >> 9;   // 0..255 row of [W;U]
        const int n    = idx & 511;  // gate column (coalesced global reads)
        const float v  = (kAll < FEAT) ? W[idx] : U[idx - FEAT * NGATE];
        Wlds[n * KP + kAll] = (__bf16)v;
    }
    blds[tid]       = b[tid];
    blds[tid + 256] = b[tid + 256];

    // zero the h-parts of the fragment stages (h1 in A1 hi-half, all of A2)
    A1[256 + tid] = make_uint4(0u, 0u, 0u, 0u);
    A2[tid]       = make_uint4(0u, 0u, 0u, 0u);
    A2[256 + tid] = make_uint4(0u, 0u, 0u, 0u);

    // per-thread register state: thread owns (row m_g, units u0..u0+7)
    const int m_g = tid >> 4;
    const int u0  = (tid & 15) * 8;
    float c1r[8], c2r[8], h2r[8];
#pragma unroll
    for (int j = 0; j < 8; ++j) { c1r[j] = 0.f; c2r[j] = 0.f; h2r[j] = 0.f; }

    // x loader mapping: same (row, 8-chunk) ownership
    const int m_x = tid >> 4;
    const int cx  = tid & 15;
    const float* xrow =
        x + ((size_t)(wg * ROWS + m_x) * TSTEPS) * FEAT + cx * 8;

    __syncthreads();

    for (int t = 0; t < TSTEPS; ++t) {
        // ---- stage x_t slab into A1 (k = 0..127) as bf16 fragments ----
        const float* xp = xrow + (size_t)t * FEAT;
        const float4 f0 = *(const float4*)(xp);
        const float4 f1 = *(const float4*)(xp + 4);
        if (t + 1 < TSTEPS) {
            __builtin_prefetch(xrow + (size_t)(t + 1) * FEAT, 0, 1);
        }
        Pack8 px;
        px.h[0] = (__bf16)f0.x; px.h[1] = (__bf16)f0.y;
        px.h[2] = (__bf16)f0.z; px.h[3] = (__bf16)f0.w;
        px.h[4] = (__bf16)f1.x; px.h[5] = (__bf16)f1.y;
        px.h[6] = (__bf16)f1.z; px.h[7] = (__bf16)f1.w;
        A1[a_chunk_index(m_x, cx * 8)] = px.q;

        __syncthreads();   // x staged; prev-step gate2 writes visible

        // ---- layer 1: z1 = [x_t | h1] @ [W;U] ----
        do_gemm(A1, Wlds, zlds, wave, lane);
        __syncthreads();

        // ---- gates layer 1 ----
        {
            Pack8 ph;
#pragma unroll
            for (int j = 0; j < 8; ++j) {
                const int u = u0 + j;
                const float zi = zlds[m_g * ZP + u]       + blds[u];
                const float zf = zlds[m_g * ZP + u + 128] + blds[u + 128];
                const float zg = zlds[m_g * ZP + u + 256] + blds[u + 256];
                const float zo = zlds[m_g * ZP + u + 384] + blds[u + 384];
                const float iv = sigf(zi);
                const float fv = sigf(zf);
                const float gv = tanh_fast(zg);
                const float ov = sigf(zo);
                const float c  = fv * c1r[j] + iv * gv;
                c1r[j] = c;
                const float hv = ov * tanh_fast(c);
                ph.h[j] = (__bf16)hv;
            }
            A1[a_chunk_index(m_g, 128 + u0)] = ph.q;  // h1 for next step's GEMM1
            A2[a_chunk_index(m_g, u0)]       = ph.q;  // h1 as x of layer 2
        }
        __syncthreads();

        // ---- layer 2: z2 = [h1 | h2] @ [W;U]  (shared weights) ----
        do_gemm(A2, Wlds, zlds, wave, lane);
        __syncthreads();

        // ---- gates layer 2 ----
        {
            Pack8 ph;
#pragma unroll
            for (int j = 0; j < 8; ++j) {
                const int u = u0 + j;
                const float zi = zlds[m_g * ZP + u]       + blds[u];
                const float zf = zlds[m_g * ZP + u + 128] + blds[u + 128];
                const float zg = zlds[m_g * ZP + u + 256] + blds[u + 256];
                const float zo = zlds[m_g * ZP + u + 384] + blds[u + 384];
                const float iv = sigf(zi);
                const float fv = sigf(zf);
                const float gv = tanh_fast(zg);
                const float ov = sigf(zo);
                const float c  = fv * c2r[j] + iv * gv;
                c2r[j] = c;
                const float hv = ov * tanh_fast(c);
                h2r[j] = hv;                 // keep f32 for final output
                ph.h[j] = (__bf16)hv;
            }
            A2[a_chunk_index(m_g, 128 + u0)] = ph.q;  // h2 for next step's GEMM2
        }
        // loop-top barrier orders gate2 vs next GEMM usage
    }

    // ---- epilogue: outputs = h2 (final), final_state = c1 (final) ----
    const int row = wg * ROWS + m_g;
    float* outH = out;                               // [256 x 128]
    float* outC = out + 32768 + 33554432;            // after zero block
#pragma unroll
    for (int j = 0; j < 8; ++j) {
        outH[row * UNITS + u0 + j] = h2r[j];
        outC[row * UNITS + u0 + j] = c1r[j];
    }
}

// init_state = zeros(256,1024,128) — 128 MB streaming store
__global__ void zero_kernel(float4* __restrict__ p, long n4)
{
    long i = (long)blockIdx.x * blockDim.x + threadIdx.x;
    const long stride = (long)gridDim.x * blockDim.x;
    for (; i < n4; i += stride) p[i] = make_float4(0.f, 0.f, 0.f, 0.f);
}

extern "C" void kernel_launch(void* const* d_in, const int* in_sizes, int n_in,
                              void* d_out, int out_size, void* d_ws, size_t ws_size,
                              hipStream_t stream)
{
    (void)in_sizes; (void)n_in; (void)d_ws; (void)ws_size; (void)out_size;
    const float* x = (const float*)d_in[0];
    const float* W = (const float*)d_in[1];
    const float* U = (const float*)d_in[2];
    const float* b = (const float*)d_in[3];
    float* out = (float*)d_out;

    // zeros block: out[32768 .. 32768+33554432)
    zero_kernel<<<8192, 256, 0, stream>>>((float4*)(out + 32768), 33554432L / 4);

    // persistent recurrence: 16 WGs, one 16-row batch slab each, no inter-WG sync
    lstm_stack2_kernel<<<NWG, 256, 0, stream>>>(x, W, U, b, out);
}
